// QLSTM_65481071395621
// MI455X (gfx1250) — compile-verified, matrix operand order checked by reading, not run on hardware
//
#include <hip/hip_runtime.h>

// ---------------- problem constants ----------------
#define T_STEPS 512
#define BATCH   64
#define DDIM    512
#define HDIM    512
#define KDIM    1024          // D + H
#define NBLK    32            // persistent workgroups; each owns 16 h-columns
#define NTHR    256           // 8 wave32 waves
#define KT_X    16            // k-tiles sourced from x
#define KT_ALL  32            // total k-tiles (K = 1024 = 32 * 32)
#define GSTR    80            // gates LDS row stride in floats (conflict-free)

// LDS layout (dynamic): [ packed W slice 128KB | gates 64x80 f32 | bias 64 f32 ]
#define W_LDS_BYTES   (4 * KT_ALL * 512 * 2)            // 131072
#define G_LDS_BYTES   (BATCH * GSTR * 4)                // 20480
#define SMEM_BYTES    (W_LDS_BYTES + G_LDS_BYTES + 64 * 4)

// workspace layout (bytes)
#define CNT_OFF   0
#define HBF0_OFF  256
#define HBF1_OFF  (256 + BATCH * HDIM * 2)
#define WPK_OFF   (256 + 2 * BATCH * HDIM * 2)
#define WPK_ELEMS (NBLK * 4 * KT_ALL * 512)             // 2,097,152 bf16
#define XBF_OFF   (WPK_OFF + WPK_ELEMS * 2)
#define XBF_ELEMS ((size_t)T_STEPS * BATCH * DDIM)      // 16,777,216 bf16
#define WS_FULL   ((size_t)XBF_OFF + XBF_ELEMS * 2)

// ---------------- tiny vector / fragment types (trivially constructible) ----
typedef __bf16  bf16x16 __attribute__((ext_vector_type(16)));
typedef float   f32x8   __attribute__((ext_vector_type(8)));
typedef float   f32x4   __attribute__((ext_vector_type(4)));
typedef unsigned u32x4  __attribute__((ext_vector_type(4)));

union Frag {
    u32x4          u[2];
    unsigned short s[16];
    bf16x16        v;
};
union Pack8 {
    unsigned short s[8];
    u32x4          u;
};

__device__ __forceinline__ unsigned short to_bf16(float f) {
    __bf16 h = (__bf16)f;                 // RNE fptrunc -> native cvt
    return __builtin_bit_cast(unsigned short, h);
}
__device__ __forceinline__ float sigmoidf_(float x) {
    return 1.0f / (1.0f + __expf(-x));
}
__device__ __forceinline__ float tanhf_(float x) {
    return 1.0f - 2.0f / (__expf(2.0f * x) + 1.0f);
}

// ---------------- init: zero h0 buffer + barrier counter ----------------
__global__ void lstm_init_kernel(unsigned short* hbf0, unsigned* cnt) {
    int i = blockIdx.x * blockDim.x + threadIdx.x;
    if (i < BATCH * HDIM) hbf0[i] = 0;
    if (i == 0) *cnt = 0u;
}

// ---------------- x fp32 -> bf16 (row-major copy), fully parallel ----------
__global__ void lstm_x2bf_kernel(const float* __restrict__ x,
                                 unsigned short* __restrict__ xbf) {
    size_t base = ((size_t)blockIdx.x * blockDim.x + threadIdx.x) * 8;
    if (base >= XBF_ELEMS) return;
    f32x4 a = *(const f32x4*)(x + base);
    f32x4 b = *(const f32x4*)(x + base + 4);
    Pack8 p;
#pragma unroll
    for (int e = 0; e < 4; ++e) {
        p.s[e]     = to_bf16(a[e]);
        p.s[e + 4] = to_bf16(b[e]);
    }
    *(u32x4*)(xbf + base) = p.u;
}

// ---------------- pack: W -> bf16, pre-swizzled into WMMA B-fragment order --
// linear index i = ((wg*4 + nt)*32 + kt)*512 + lane*16 + e
// fragment element (lane, e) holds B[k][n] = W_gate[n][k] with
//   n  = wg*16 + (lane & 15)
//   k  = kt*32 + (e>>3)*16 + (lane>>4)*8 + (e&7)
__global__ void lstm_pack_kernel(const float* __restrict__ Wf,
                                 const float* __restrict__ Wi,
                                 const float* __restrict__ Wg,
                                 const float* __restrict__ Wo,
                                 unsigned short* __restrict__ wpack) {
    int i = blockIdx.x * blockDim.x + threadIdx.x;
    if (i >= WPK_ELEMS) return;
    int e    = i & 15;
    int lane = (i >> 4) & 31;
    int f    = i >> 9;
    int kt   = f & 31;
    int nt   = (f >> 5) & 3;
    int wg   = f >> 7;
    int hi   = lane >> 4;
    int k    = kt * 32 + (e >> 3) * 16 + hi * 8 + (e & 7);
    int n    = wg * 16 + (lane & 15);
    const float* Wsrc = (nt == 0) ? Wf : (nt == 1) ? Wi : (nt == 2) ? Wg : Wo;
    wpack[i] = to_bf16(Wsrc[(size_t)n * KDIM + k]);
}

// ---------------- persistent scan kernel ----------------
template <bool USE_XBF>
__global__ void __launch_bounds__(NTHR, 1)
lstm_scan_kernel(const float* __restrict__ x,
                 const unsigned short* __restrict__ xbf,
                 const unsigned short* __restrict__ wpack,
                 unsigned short* __restrict__ hbf0,
                 unsigned short* __restrict__ hbf1,
                 const float* __restrict__ bf_, const float* __restrict__ bi_,
                 const float* __restrict__ bg_, const float* __restrict__ bo_,
                 float* __restrict__ out,
                 unsigned* __restrict__ cnt) {
    extern __shared__ char smem[];
    unsigned short* wlds  = (unsigned short*)smem;
    float*          gates = (float*)(smem + W_LDS_BYTES);
    float*          biasl = (float*)(smem + W_LDS_BYTES + G_LDS_BYTES);

    const int wg   = blockIdx.x;        // owns h-columns [wg*16, wg*16+16)
    const int tid  = threadIdx.x;
    const int wv   = tid >> 5;          // wave32
    const int lane = tid & 31;
    const int mt   = wv & 3;            // batch-row tile (4 x 16 rows)
    const int grp  = wv >> 2;           // 0 -> {f,i} tiles, 1 -> {g,o} tiles
    const int hi   = lane >> 4;
    const int nl   = lane & 15;

    // stage this workgroup's packed W slice into LDS (read from HBM once)
    {
        const u32x4* src = (const u32x4*)(wpack + (size_t)wg * (4 * KT_ALL * 512));
        u32x4*       dst = (u32x4*)wlds;
        for (int i = tid; i < W_LDS_BYTES / 16; i += NTHR) dst[i] = src[i];
    }
    if (tid < 64) {
        int g = tid >> 4, c = tid & 15;
        const float* bp = (g == 0) ? bf_ : (g == 1) ? bi_ : (g == 2) ? bg_ : bo_;
        biasl[tid] = bp[wg * 16 + c];
    }
    __syncthreads();

    const int nt0 = grp * 2, nt1 = grp * 2 + 1;
    const unsigned short* b0base = wlds + nt0 * KT_ALL * 512 + lane * 16;
    const unsigned short* b1base = wlds + nt1 * KT_ALL * 512 + lane * 16;
    const int mrow = mt * 16 + nl;          // batch row for A fragments / C cols
    const int gc0  = nt0 * 16 + nl;         // C-tile column in gates buffer
    const int gc1  = nt1 * 16 + nl;

    // elementwise slice: thread -> (row, 4 h-cols)
    const int erow = tid & 63;
    const int ecg  = tid >> 6;
    float cst[4] = {0.f, 0.f, 0.f, 0.f};

    for (int t = 0; t < T_STEPS; ++t) {
        const unsigned short* hread  = (t & 1) ? hbf1 : hbf0;  // h_{t-1}
        unsigned short*       hwrite = (t & 1) ? hbf0 : hbf1;  // h_t
        const float*          xrow  = x   + ((size_t)t * BATCH + mrow) * DDIM;
        const unsigned short* xbrow = xbf + ((size_t)t * BATCH + mrow) * DDIM;
        const unsigned short* hrow  = hread + (size_t)mrow * HDIM;
        if (t + 1 < T_STEPS) {
            if (USE_XBF) __builtin_prefetch(xbrow + (size_t)BATCH * DDIM, 0, 0);
            else         __builtin_prefetch(xrow  + (size_t)BATCH * DDIM, 0, 0);
        }

        f32x8 acc0 = {0.f, 0.f, 0.f, 0.f, 0.f, 0.f, 0.f, 0.f};
        f32x8 acc1 = {0.f, 0.f, 0.f, 0.f, 0.f, 0.f, 0.f, 0.f};

        // K-tiles from x
        for (int kt = 0; kt < KT_X; ++kt) {
            Frag a;
            if (USE_XBF) {
                const unsigned short* p = xbrow + kt * 32 + hi * 8;
                a.u[0] = *(const u32x4*)(p);
                a.u[1] = *(const u32x4*)(p + 16);
            } else {
                const float* p = xrow + kt * 32 + hi * 8;
                f32x4 f0 = *(const f32x4*)(p);
                f32x4 f1 = *(const f32x4*)(p + 4);
                f32x4 f2 = *(const f32x4*)(p + 16);
                f32x4 f3 = *(const f32x4*)(p + 20);
#pragma unroll
                for (int e = 0; e < 4; ++e) {
                    a.v[e]      = (__bf16)f0[e];
                    a.v[e + 4]  = (__bf16)f1[e];
                    a.v[e + 8]  = (__bf16)f2[e];
                    a.v[e + 12] = (__bf16)f3[e];
                }
            }
            Frag b0, b1;
            const u32x4* q0 = (const u32x4*)(b0base + kt * 512);
            const u32x4* q1 = (const u32x4*)(b1base + kt * 512);
            b0.u[0] = q0[0]; b0.u[1] = q0[1];
            b1.u[0] = q1[0]; b1.u[1] = q1[1];
            acc0 = __builtin_amdgcn_wmma_f32_16x16x32_bf16(false, a.v, false, b0.v,
                                                           (short)0, acc0, false, false);
            acc1 = __builtin_amdgcn_wmma_f32_16x16x32_bf16(false, a.v, false, b1.v,
                                                           (short)0, acc1, false, false);
        }
        // K-tiles from h_{t-1} (already bf16)
        for (int kt = KT_X; kt < KT_ALL; ++kt) {
            const unsigned short* p = hrow + (kt - KT_X) * 32 + hi * 8;
            Frag a;
            a.u[0] = *(const u32x4*)(p);
            a.u[1] = *(const u32x4*)(p + 16);
            Frag b0, b1;
            const u32x4* q0 = (const u32x4*)(b0base + kt * 512);
            const u32x4* q1 = (const u32x4*)(b1base + kt * 512);
            b0.u[0] = q0[0]; b0.u[1] = q0[1];
            b1.u[0] = q1[0]; b1.u[1] = q1[1];
            acc0 = __builtin_amdgcn_wmma_f32_16x16x32_bf16(false, a.v, false, b0.v,
                                                           (short)0, acc0, false, false);
            acc1 = __builtin_amdgcn_wmma_f32_16x16x32_bf16(false, a.v, false, b1.v,
                                                           (short)0, acc1, false, false);
        }

        // C/D layout: vgpr r -> row (mt*16 + hi*8 + r), col (nt*16 + (lane&15))
#pragma unroll
        for (int r = 0; r < 8; ++r) {
            int grow = mt * 16 + hi * 8 + r;
            gates[grow * GSTR + gc0] = acc0[r];
            gates[grow * GSTR + gc1] = acc1[r];
        }
        __syncthreads();

        // elementwise LSTM cell update for this workgroup's 16 h-columns
        const float* gp = gates + erow * GSTR;
#pragma unroll
        for (int j = 0; j < 4; ++j) {
            int col = ecg * 4 + j;
            float fg = sigmoidf_(gp[col]      + biasl[col]);
            float ig = sigmoidf_(gp[16 + col] + biasl[16 + col]);
            float gg = tanhf_   (gp[32 + col] + biasl[32 + col]);
            float og = sigmoidf_(gp[48 + col] + biasl[48 + col]);
            float cn = fg * cst[j] + ig * gg;
            cst[j] = cn;
            float hn = og * tanhf_(cn);
            int hcol = wg * 16 + col;
            out[((size_t)t * BATCH + erow) * HDIM + hcol] = hn;
            hwrite[(size_t)erow * HDIM + hcol] = to_bf16(hn);
            if (t == T_STEPS - 1)  // hT output region
                out[(size_t)T_STEPS * BATCH * HDIM + (size_t)erow * HDIM + hcol] = hn;
        }

        // device-wide barrier (monotonic counter; all 32 blocks co-resident)
        __threadfence();
        __syncthreads();
        if (tid == 0) {
            __hip_atomic_fetch_add(cnt, 1u, __ATOMIC_ACQ_REL, __HIP_MEMORY_SCOPE_AGENT);
            const unsigned target = (unsigned)(t + 1) * NBLK;
            while (__hip_atomic_load(cnt, __ATOMIC_ACQUIRE, __HIP_MEMORY_SCOPE_AGENT) < target)
                __builtin_amdgcn_s_sleep(2);
        }
        __syncthreads();
        __threadfence();
    }

    // cT output region
    float* cT = out + (size_t)T_STEPS * BATCH * HDIM + (size_t)BATCH * HDIM;
#pragma unroll
    for (int j = 0; j < 4; ++j) {
        int hcol = wg * 16 + ecg * 4 + j;
        cT[(size_t)erow * HDIM + hcol] = cst[j];
    }
}

// ---------------- host entry ----------------
extern "C" void kernel_launch(void* const* d_in, const int* in_sizes, int n_in,
                              void* d_out, int out_size, void* d_ws, size_t ws_size,
                              hipStream_t stream) {
    (void)in_sizes; (void)n_in; (void)out_size;
    const float* x  = (const float*)d_in[0];
    const float* Wf = (const float*)d_in[1];
    const float* bf = (const float*)d_in[2];
    const float* Wi = (const float*)d_in[3];
    const float* bi = (const float*)d_in[4];
    const float* Wg = (const float*)d_in[5];
    const float* bg = (const float*)d_in[6];
    const float* Wo = (const float*)d_in[7];
    const float* bo = (const float*)d_in[8];
    float* out = (float*)d_out;

    char* ws = (char*)d_ws;
    unsigned*       cnt   = (unsigned*)(ws + CNT_OFF);
    unsigned short* hbf0  = (unsigned short*)(ws + HBF0_OFF);
    unsigned short* hbf1  = (unsigned short*)(ws + HBF1_OFF);
    unsigned short* wpack = (unsigned short*)(ws + WPK_OFF);
    unsigned short* xbf   = (unsigned short*)(ws + XBF_OFF);

    lstm_init_kernel<<<(BATCH * HDIM + NTHR - 1) / NTHR, NTHR, 0, stream>>>(hbf0, cnt);
    lstm_pack_kernel<<<(WPK_ELEMS + NTHR - 1) / NTHR, NTHR, 0, stream>>>(Wf, Wi, Wg, Wo, wpack);

    const bool use_xbf = (ws_size >= WS_FULL);   // same every call -> deterministic
    if (use_xbf) {
        lstm_x2bf_kernel<<<(int)(XBF_ELEMS / 8 / NTHR), NTHR, 0, stream>>>(x, xbf);
        (void)hipFuncSetAttribute(reinterpret_cast<const void*>(lstm_scan_kernel<true>),
                                  hipFuncAttributeMaxDynamicSharedMemorySize, SMEM_BYTES);
        lstm_scan_kernel<true><<<NBLK, NTHR, SMEM_BYTES, stream>>>(
            x, xbf, wpack, hbf0, hbf1, bf, bi, bg, bo, out, cnt);
    } else {
        (void)hipFuncSetAttribute(reinterpret_cast<const void*>(lstm_scan_kernel<false>),
                                  hipFuncAttributeMaxDynamicSharedMemorySize, SMEM_BYTES);
        lstm_scan_kernel<false><<<NBLK, NTHR, SMEM_BYTES, stream>>>(
            x, xbf, wpack, hbf0, hbf1, bf, bi, bg, bo, out, cnt);
    }
}
